// SparseDoubleConv_25598005084695
// MI455X (gfx1250) — matrix operand order repaired and checked by reading, not compiled, not run on hardware
//
#include <hip/hip_runtime.h>

// ---------------------------------------------------------------------------
// SparseDoubleConv for MI455X (gfx1250): rulebook gather + bf16 WMMA GEMM,
// packed weights staged in LDS via GLOBAL_LOAD_ASYNC_TO_LDS_B128 (ASYNCcnt),
// deterministic two-pass BatchNorm + LeakyReLU, all scratch in d_ws.
// ---------------------------------------------------------------------------

typedef __attribute__((ext_vector_type(16))) __bf16 v16bf;
typedef __attribute__((ext_vector_type(8)))  __bf16 bf16x8;
typedef __attribute__((ext_vector_type(8)))  float  v8f;
typedef __attribute__((ext_vector_type(4)))  int    i32x4;

#define NSITES   200000
#define CIN      32
#define COUT     64
#define NK       27
#define NTILES   (NSITES / 16)       // 12500 (N is a multiple of 16)
#define WAVES_PB 8                   // 256-thread blocks = 8 wave32
#define STAT_BLK 256                 // blocks in the BN stats pass
#define NEG_SLOPE 0.05f
#define BN_EPS    1e-4f

#define W1P_BYTES (NK * 4 * 32 * 16 * 2)      // 110592 packed bf16 bytes
#define W2P_BYTES (NK * 8 * 32 * 16 * 2)      // 221184 packed bf16 bytes

// ---- CDNA5 async global->LDS (ASYNCcnt) with safe fallback ---------------
#if defined(__has_builtin)
#  if __has_builtin(__builtin_amdgcn_global_load_async_to_lds_b128) && \
      __has_builtin(__builtin_amdgcn_s_wait_asynccnt)
#    define USE_ASYNC_LDS 1
#  endif
#endif
#ifndef USE_ASYNC_LDS
#  define USE_ASYNC_LDS 0
#endif

#define AS_GLOBAL __attribute__((address_space(1)))
#define AS_LDS    __attribute__((address_space(3)))

typedef AS_GLOBAL i32x4 gvec4_t;   // global-AS int4
typedef AS_LDS    i32x4 lvec4_t;   // LDS-AS int4

// Stage nbytes (multiple of 16) from global into LDS, block-cooperatively.
// Ends with a full-block barrier.
__device__ __forceinline__ void stage_to_lds(const char* __restrict__ gsrc,
                                             char* __restrict__ ldst, int nbytes) {
#if USE_ASYNC_LDS
    for (int i = (int)threadIdx.x * 16; i < nbytes; i += 256 * 16) {
        __builtin_amdgcn_global_load_async_to_lds_b128(
            (gvec4_t*)(uintptr_t)(gsrc + i),
            (lvec4_t*)(uint32_t)(uintptr_t)(ldst + i),
            0, 0);
    }
    __builtin_amdgcn_s_wait_asynccnt(0);
    __syncthreads();
#else
    const uint4* g4 = (const uint4*)gsrc;
    uint4* l4 = (uint4*)ldst;
    for (int i = (int)threadIdx.x; i < nbytes / 16; i += 256) l4[i] = g4[i];
    __syncthreads();
#endif
}

// -------------------------------------------------------------------- utils
__device__ __forceinline__ float lrelu(float y) {
    return y > 0.0f ? y : NEG_SLOPE * y;
}

// ---------------------------------------------------------- weight packing
// Packed B-fragment layout (V_WMMA_F32_16X16X32_BF16, B is K x 16, col-major
// per lane): lane = 16*half + n,  n = N column (0..15),
//   K = 8*half + slot            (slot 0..7)
//   K = 16 + 8*half + (slot - 8) (slot 8..15)
// Flat packed index: (((k*T + t)*32) + lane)*16 + slot, 16 bf16 per lane
// so each lane reads 32 contiguous bytes (two b128 loads).
__global__ void pack_w1(const float* __restrict__ W1, __bf16* __restrict__ Wp) {
    int i = blockIdx.x * 256 + threadIdx.x;
    if (i >= NK * 4 * 32 * 16) return;
    int slot =  i        & 15;
    int lane = (i >> 4)  & 31;
    int t    = (i >> 9)  & 3;
    int k    =  i >> 11;
    int n = t * 16 + (lane & 15);
    int K = (slot < 8) ? ((lane >> 4) * 8 + slot) : (16 + (lane >> 4) * 8 + (slot - 8));
    Wp[i] = (__bf16)W1[(k * CIN + K) * COUT + n];
}

__global__ void pack_w2(const float* __restrict__ W2, __bf16* __restrict__ Wp) {
    int i = blockIdx.x * 256 + threadIdx.x;
    if (i >= NK * 2 * 4 * 32 * 16) return;
    int slot =  i        & 15;
    int lane = (i >> 4)  & 31;
    int t    = (i >> 9)  & 3;
    int s    = (i >> 11) & 1;                 // K slab: 0..31 / 32..63
    int k    =  i >> 12;
    int n = t * 16 + (lane & 15);
    int K = s * 32 + ((slot < 8) ? ((lane >> 4) * 8 + slot) : (16 + (lane >> 4) * 8 + (slot - 8)));
    Wp[i] = (__bf16)W2[(k * COUT + K) * COUT + n];
}

// f32 features -> bf16 table with a trailing zero row (gather sentinel = N).
__global__ void cvt_feat_bf16(const float* __restrict__ F, __bf16* __restrict__ Fp) {
    const long total = (long)NSITES * CIN;
    long stride = (long)gridDim.x * blockDim.x;
    for (long i = (long)blockIdx.x * blockDim.x + threadIdx.x;
         i < total + CIN; i += stride) {
        Fp[i] = (i < total) ? (__bf16)F[i] : (__bf16)0.0f;
    }
}

// ------------------------------------------------------------------ conv 1
// One wave per 16-site tile; A = 16 gathered bf16 rows (K=32), B = packed
// weight fragments read from LDS, C/D = 4 x v8f f32 accumulators.
__global__ __launch_bounds__(256)
void conv1_wmma(const __bf16* __restrict__ featp, const int* __restrict__ nbr,
                const __bf16* __restrict__ Wp, float* __restrict__ X1) {
    extern __shared__ char smem[];
    stage_to_lds((const char*)Wp, smem, W1P_BYTES);

    int wave = threadIdx.x >> 5;
    int lane = threadIdx.x & 31;
    int tile = blockIdx.x * WAVES_PB + wave;
    if (tile >= NTILES) return;                 // uniform per wave: EXEC stays full
    int row  = lane & 15;
    int half = lane >> 4;
    const int* nrow = nbr + (size_t)(tile * 16 + row) * NK;

    v8f c0 = {}, c1 = {}, c2 = {}, c3 = {};
    for (int k = 0; k < NK; ++k) {
        int j = nrow[k];                        // sentinel NSITES -> zero row
        const bf16x8* fr = (const bf16x8*)(featp + (size_t)j * CIN) + half;
        union { v16bf v; bf16x8 h[2]; } a;
        a.h[0] = fr[0];                         // K = half*8      .. +7
        a.h[1] = fr[2];                         // K = 16 + half*8 .. +7
        const v16bf* bp = (const v16bf*)smem + (size_t)(k * 4) * 32 + lane;
        c0 = __builtin_amdgcn_wmma_f32_16x16x32_bf16(false, a.v, false, bp[0],
                                                     (short)0, c0, false, false);
        c1 = __builtin_amdgcn_wmma_f32_16x16x32_bf16(false, a.v, false, bp[32],
                                                     (short)0, c1, false, false);
        c2 = __builtin_amdgcn_wmma_f32_16x16x32_bf16(false, a.v, false, bp[64],
                                                     (short)0, c2, false, false);
        c3 = __builtin_amdgcn_wmma_f32_16x16x32_bf16(false, a.v, false, bp[96],
                                                     (short)0, c3, false, false);
    }
    // C/D layout: VGPR r, lanes 0-15: (M=r, N=lane); lanes 16-31: (M=8+r, N=lane-16)
    float* o = X1 + ((size_t)tile * 16 + half * 8) * COUT + row;
#pragma unroll
    for (int r = 0; r < 8; ++r) {
        float* p = o + (size_t)r * COUT;
        p[0] = c0[r]; p[16] = c1[r]; p[32] = c2[r]; p[48] = c3[r];
    }
}

// ------------------------------------------------------------------ conv 2
// Same, but Cin=64 -> two K=32 WMMAs per rulebook offset.
__global__ __launch_bounds__(256)
void conv2_wmma(const __bf16* __restrict__ Xn, const int* __restrict__ nbr,
                const __bf16* __restrict__ Wp, float* __restrict__ X2) {
    extern __shared__ char smem[];
    stage_to_lds((const char*)Wp, smem, W2P_BYTES);

    int wave = threadIdx.x >> 5;
    int lane = threadIdx.x & 31;
    int tile = blockIdx.x * WAVES_PB + wave;
    if (tile >= NTILES) return;
    int row  = lane & 15;
    int half = lane >> 4;
    const int* nrow = nbr + (size_t)(tile * 16 + row) * NK;

    v8f c0 = {}, c1 = {}, c2 = {}, c3 = {};
    for (int k = 0; k < NK; ++k) {
        int j = nrow[k];
        const bf16x8* fr = (const bf16x8*)(Xn + (size_t)j * COUT) + half;
        union { v16bf v; bf16x8 h[2]; } a0, a1;
        a0.h[0] = fr[0]; a0.h[1] = fr[2];       // K slab 0..31
        a1.h[0] = fr[4]; a1.h[1] = fr[6];       // K slab 32..63
        const v16bf* bp = (const v16bf*)smem + (size_t)(k * 8) * 32 + lane;
        c0 = __builtin_amdgcn_wmma_f32_16x16x32_bf16(false, a0.v, false, bp[0],
                                                     (short)0, c0, false, false);
        c1 = __builtin_amdgcn_wmma_f32_16x16x32_bf16(false, a0.v, false, bp[32],
                                                     (short)0, c1, false, false);
        c2 = __builtin_amdgcn_wmma_f32_16x16x32_bf16(false, a0.v, false, bp[64],
                                                     (short)0, c2, false, false);
        c3 = __builtin_amdgcn_wmma_f32_16x16x32_bf16(false, a0.v, false, bp[96],
                                                     (short)0, c3, false, false);
        c0 = __builtin_amdgcn_wmma_f32_16x16x32_bf16(false, a1.v, false, bp[128],
                                                     (short)0, c0, false, false);
        c1 = __builtin_amdgcn_wmma_f32_16x16x32_bf16(false, a1.v, false, bp[160],
                                                     (short)0, c1, false, false);
        c2 = __builtin_amdgcn_wmma_f32_16x16x32_bf16(false, a1.v, false, bp[192],
                                                     (short)0, c2, false, false);
        c3 = __builtin_amdgcn_wmma_f32_16x16x32_bf16(false, a1.v, false, bp[224],
                                                     (short)0, c3, false, false);
    }
    float* o = X2 + ((size_t)tile * 16 + half * 8) * COUT + row;
#pragma unroll
    for (int r = 0; r < 8; ++r) {
        float* p = o + (size_t)r * COUT;
        p[0] = c0[r]; p[16] = c1[r]; p[32] = c2[r]; p[48] = c3[r];
    }
}

// ---------------------------------------------------- BatchNorm (two-pass)
// Pass 1: per-block partial sum / sum-of-squares per channel (deterministic,
// no float atomics). partials layout: [block][2][64].
__global__ __launch_bounds__(256)
void bn_stats(const float* __restrict__ X, float* __restrict__ partials) {
    int ch  = threadIdx.x & 63;
    int sub = threadIdx.x >> 6;                 // 0..3 row-streams per block
    float s = 0.0f, q = 0.0f;
    int rstride = gridDim.x * 4;
    for (int r = blockIdx.x * 4 + sub; r < NSITES; r += rstride) {
        float v = X[(size_t)r * COUT + ch];
        s += v; q += v * v;
    }
    __shared__ float ls[4][64], lq[4][64];
    ls[sub][ch] = s; lq[sub][ch] = q;
    __syncthreads();
    if (threadIdx.x < 64) {
        int c = threadIdx.x;
        float S = ls[0][c] + ls[1][c] + ls[2][c] + ls[3][c];
        float Q = lq[0][c] + lq[1][c] + lq[2][c] + lq[3][c];
        partials[(size_t)blockIdx.x * 128 + c]      = S;
        partials[(size_t)blockIdx.x * 128 + 64 + c] = Q;
    }
}

// Pass 2: 64 threads fold partials -> scale/shift (ss[0..63]=scale, [64..127]=shift).
__global__ void bn_finalize(const float* __restrict__ partials,
                            const float* __restrict__ g, const float* __restrict__ b,
                            float* __restrict__ ss) {
    int ch = threadIdx.x;
    if (ch >= 64) return;
    float S = 0.0f, Q = 0.0f;
    for (int i = 0; i < STAT_BLK; ++i) {
        S += partials[(size_t)i * 128 + ch];
        Q += partials[(size_t)i * 128 + 64 + ch];
    }
    float inv_n = 1.0f / (float)NSITES;
    float mean = S * inv_n;
    float var  = Q * inv_n - mean * mean;       // biased variance (matches ref)
    float sc = g[ch] * rsqrtf(var + BN_EPS);
    ss[ch]      = sc;
    ss[64 + ch] = b[ch] - mean * sc;
}

// Apply BN + LeakyReLU -> bf16 (layer-2 input, with trailing zero pad row).
__global__ void bn_apply_bf16(const float* __restrict__ X, const float* __restrict__ ss,
                              __bf16* __restrict__ Y) {
    const long total = (long)NSITES * COUT;
    long stride = (long)gridDim.x * blockDim.x;
    for (long i = (long)blockIdx.x * blockDim.x + threadIdx.x;
         i < total + COUT; i += stride) {
        if (i < total) {
            int ch = (int)(i & 63);
            Y[i] = (__bf16)lrelu(X[i] * ss[ch] + ss[64 + ch]);
        } else {
            Y[i] = (__bf16)0.0f;                // sentinel row
        }
    }
}

// Apply BN + LeakyReLU -> f32 final output.
__global__ void bn_apply_f32(const float* __restrict__ X, const float* __restrict__ ss,
                             float* __restrict__ Y) {
    const long total = (long)NSITES * COUT;
    long stride = (long)gridDim.x * blockDim.x;
    for (long i = (long)blockIdx.x * blockDim.x + threadIdx.x; i < total; i += stride) {
        int ch = (int)(i & 63);
        Y[i] = lrelu(X[i] * ss[ch] + ss[64 + ch]);
    }
}

// ------------------------------------------------------------------- launch
extern "C" void kernel_launch(void* const* d_in, const int* in_sizes, int n_in,
                              void* d_out, int out_size, void* d_ws, size_t ws_size,
                              hipStream_t stream) {
    const float* features = (const float*)d_in[0];
    const int*   nbr      = (const int*)d_in[1];
    const float* W1       = (const float*)d_in[2];
    const float* g1       = (const float*)d_in[3];
    const float* b1       = (const float*)d_in[4];
    const float* W2       = (const float*)d_in[5];
    const float* g2       = (const float*)d_in[6];
    const float* b2       = (const float*)d_in[7];
    float* out = (float*)d_out;

    // workspace carve-up (256-B aligned)
    char* ws = (char*)d_ws;
    size_t off = 0;
    auto take = [&](size_t bytes) { char* p = ws + off; off = (off + bytes + 255) & ~(size_t)255; return p; };
    __bf16* W1p   = (__bf16*)take((size_t)W1P_BYTES);                               // 108 KB
    __bf16* W2p   = (__bf16*)take((size_t)W2P_BYTES);                               // 216 KB
    __bf16* featp = (__bf16*)take((size_t)(NSITES + 1) * CIN * sizeof(__bf16));     // 12.8 MB
    __bf16* Xn    = (__bf16*)take((size_t)(NSITES + 1) * COUT * sizeof(__bf16));    // 25.6 MB
    float*  Xtmp  = (float*) take((size_t)NSITES * COUT * sizeof(float));           // 51.2 MB
    float*  part  = (float*) take((size_t)STAT_BLK * 128 * sizeof(float));          // 128 KB
    float*  ss1   = (float*) take(128 * sizeof(float));
    float*  ss2   = (float*) take(128 * sizeof(float));
    (void)in_sizes; (void)n_in; (void)out_size; (void)ws_size;

    const int convBlocks = (NTILES + WAVES_PB - 1) / WAVES_PB;   // 1563

    // prep
    pack_w1<<<(NK * 4 * 32 * 16 + 255) / 256, 256, 0, stream>>>(W1, W1p);
    pack_w2<<<(NK * 2 * 4 * 32 * 16 + 255) / 256, 256, 0, stream>>>(W2, W2p);
    cvt_feat_bf16<<<1024, 256, 0, stream>>>(features, featp);

    // layer 1
    conv1_wmma<<<convBlocks, 256, W1P_BYTES, stream>>>(featp, nbr, W1p, Xtmp);
    bn_stats<<<STAT_BLK, 256, 0, stream>>>(Xtmp, part);
    bn_finalize<<<1, 64, 0, stream>>>(part, g1, b1, ss1);
    bn_apply_bf16<<<2048, 256, 0, stream>>>(Xtmp, ss1, Xn);

    // layer 2
    conv2_wmma<<<convBlocks, 256, W2P_BYTES, stream>>>(Xn, nbr, W2p, Xtmp);
    bn_stats<<<STAT_BLK, 256, 0, stream>>>(Xtmp, part);
    bn_finalize<<<1, 64, 0, stream>>>(part, g2, b2, ss2);
    bn_apply_f32<<<2048, 256, 0, stream>>>(Xtmp, ss2, out);
}
